// PrefixLinearAttention_79310866088615
// MI455X (gfx1250) — matrix operand order repaired and checked by reading, not compiled, not run on hardware
//
#include <hip/hip_runtime.h>

// Problem constants (match reference)
#define B_   2
#define N_   2048
#define D_   1024
#define H_   16
#define FD_  16
#define HD_  64
#define ENC_ 512
#define FP_  288   // feature dim 273 padded to 288 = 9 * 32 (zero pad => exact result)

typedef __attribute__((ext_vector_type(16))) __bf16 v16bf;
typedef __attribute__((ext_vector_type(8)))  __bf16 v8bf;
typedef __attribute__((ext_vector_type(8)))  float  v8f;
typedef int v4i_ __attribute__((vector_size(4 * sizeof(int))));

static __device__ __forceinline__ v8f zero8() {
  v8f z;
#pragma unroll
  for (int e = 0; e < 8; ++e) z[e] = 0.f;
  return z;
}

static __device__ __forceinline__ v16bf pack16(v8bf lo, v8bf hi) {
  v16bf r;
#pragma unroll
  for (int e = 0; e < 8; ++e) { r[e] = lo[e]; r[8 + e] = hi[e]; }
  return r;
}

// A-operand (16x32 bf16), memory row-major: elem(M,K) at base[M*ld + K].
// ISA layout: lanes 0-15 rows M, holding K 0-7 & 16-23; lanes 16-31 K 8-15 & 24-31.
static __device__ __forceinline__ v16bf load_a_rm(const __bf16* base, int ld) {
  int lane = (int)(threadIdx.x & 31u);
  const __bf16* p = base + (size_t)(lane & 15) * ld + ((lane >> 4) << 3);
  v8bf lo = *reinterpret_cast<const v8bf*>(p);
  v8bf hi = *reinterpret_cast<const v8bf*>(p + 16);
  return pack16(lo, hi);
}

// B-operand (32x16 bf16), memory holds B^T row-major: elem(K,N) at base[N*ld + K].
// ISA layout: lane = column; lanes 0-15 K 0-7 & 16-23; lanes 16-31 K 8-15 & 24-31.
static __device__ __forceinline__ v16bf load_b_tr(const __bf16* base, int ld) {
  int lane = (int)(threadIdx.x & 31u);
  const __bf16* p = base + (size_t)(lane & 15) * ld + ((lane >> 4) << 3);
  v8bf lo = *reinterpret_cast<const v8bf*>(p);
  v8bf hi = *reinterpret_cast<const v8bf*>(p + 16);
  return pack16(lo, hi);
}

static __device__ __forceinline__ v8f wmma_bf16(v16bf a, v16bf b, v8f c) {
  return __builtin_amdgcn_wmma_f32_16x16x32_bf16(false, a, false, b, (short)0, c,
                                                 false, false);
}

// ---------------------------------------------------------------------------
// gfx1250 async global->LDS copy (ASYNCcnt tracked), guarded so either
// toolchain still compiles; fallback is a plain load + ds_store.
#if __has_builtin(__builtin_amdgcn_global_load_async_to_lds_b128)
#define ASYNC_COPY 1
#else
#define ASYNC_COPY 0
#endif

static __device__ __forceinline__ void cp16(__bf16* lds_dst, const __bf16* gsrc) {
#if ASYNC_COPY
  __builtin_amdgcn_global_load_async_to_lds_b128(
      (__attribute__((address_space(1))) v4i_*)(uintptr_t)gsrc,
      (__attribute__((address_space(3))) v4i_*)(unsigned)(uintptr_t)lds_dst,
      0, 0);
#else
  *reinterpret_cast<v8bf*>(lds_dst) = *reinterpret_cast<const v8bf*>(gsrc);
#endif
}

template <int Nw>
static __device__ __forceinline__ void wait_async() {
#if ASYNC_COPY
#if __has_builtin(__builtin_amdgcn_s_wait_asynccnt)
  __builtin_amdgcn_s_wait_asynccnt(Nw);
#else
  asm volatile("s_wait_asynccnt %0" ::"i"(Nw) : "memory");
#endif
#endif
}

// ---------------------------------------------------------------------------
// Elementwise f32 -> bf16 convert
__global__ void f2bf_kernel(const float* __restrict__ in, __bf16* __restrict__ out,
                            int n) {
  int i = blockIdx.x * blockDim.x + threadIdx.x;
  if (i < n) out[i] = (__bf16)in[i];
}

// ---------------------------------------------------------------------------
// Generic GEMM: C(MxNout, f32) = A(MxK, bf16 row-major, with row remap) @ W^T,
// W is (Nout x K, bf16 row-major). Row remap: row -> (row/segLen)*segStride +
// row%segLen (gathers x[:, :ENC] rows for the encoder projections).
// Block = 256 thr = 8 waves; wave tile 32x64; block tile 64x256.
__global__ __launch_bounds__(256) void gemm_xwt(
    const __bf16* __restrict__ A, const __bf16* __restrict__ W,
    float* __restrict__ C, int M, int Nout, int K, int segLen, int segStride) {
  int wave = (int)(threadIdx.x >> 5);
  int lane = (int)(threadIdx.x & 31u);
  int wm = wave & 1;   // 2 waves along M
  int wn = wave >> 1;  // 4 waves along N
  int mb = blockIdx.y * 64 + wm * 32;
  int nb = blockIdx.x * 256 + wn * 64;
  if (mb >= M || nb >= Nout) return;

  const __bf16* aptr[2];
#pragma unroll
  for (int i = 0; i < 2; ++i) {
    int row = mb + i * 16 + (lane & 15);
    row = (row / segLen) * segStride + (row % segLen);
    aptr[i] = A + (size_t)row * K + ((lane >> 4) << 3);
  }
  const __bf16* bptr[4];
#pragma unroll
  for (int j = 0; j < 4; ++j)
    bptr[j] = W + (size_t)(nb + j * 16 + (lane & 15)) * K + ((lane >> 4) << 3);

  v8f acc[2][4];
#pragma unroll
  for (int i = 0; i < 2; ++i)
#pragma unroll
    for (int j = 0; j < 4; ++j) acc[i][j] = zero8();

  for (int kk = 0; kk < K; kk += 32) {
    v16bf a[2], b[4];
#pragma unroll
    for (int i = 0; i < 2; ++i) {
      v8bf lo = *reinterpret_cast<const v8bf*>(aptr[i] + kk);
      v8bf hi = *reinterpret_cast<const v8bf*>(aptr[i] + kk + 16);
      a[i] = pack16(lo, hi);
    }
#pragma unroll
    for (int j = 0; j < 4; ++j) {
      v8bf lo = *reinterpret_cast<const v8bf*>(bptr[j] + kk);
      v8bf hi = *reinterpret_cast<const v8bf*>(bptr[j] + kk + 16);
      b[j] = pack16(lo, hi);
    }
#pragma unroll
    for (int i = 0; i < 2; ++i)
#pragma unroll
      for (int j = 0; j < 4; ++j) acc[i][j] = wmma_bf16(a[i], b[j], acc[i][j]);
  }

  int half = lane >> 4, cn = lane & 15;
#pragma unroll
  for (int i = 0; i < 2; ++i)
#pragma unroll
    for (int j = 0; j < 4; ++j) {
      float* cp = C + (size_t)(mb + i * 16 + 8 * half) * Nout + (nb + j * 16 + cn);
#pragma unroll
      for (int r = 0; r < 8; ++r) cp[(size_t)r * Nout] = acc[i][j][r];
    }
}

// ---------------------------------------------------------------------------
// Taylor feature map: per (token, head), 16 f32 values -> 288 bf16 features
// [1, x/2, outer(x,x)/(4*sqrt(2)), 0-pad].
__global__ void feature_kernel(const float* __restrict__ g, __bf16* __restrict__ f,
                               int Ntok) {
  int idx = blockIdx.x * blockDim.x + threadIdx.x;
  int total = B_ * Ntok * H_;
  if (idx >= total) return;
  int h = idx & (H_ - 1);
  int m = idx >> 4;  // token index in 0..B*Ntok-1
  int b = m / Ntok, n = m % Ntok;
  const float* src = g + (size_t)m * (H_ * FD_) + h * FD_;
  float v[16];
#pragma unroll
  for (int i = 0; i < 16; ++i) v[i] = src[i];
  __bf16* dst = f + ((size_t)(b * H_ + h) * Ntok + n) * FP_;
  dst[0] = (__bf16)1.0f;
#pragma unroll
  for (int i = 0; i < 16; ++i) dst[1 + i] = (__bf16)(v[i] * 0.5f);
  const float s = 0.17677669529663687f;  // 1/(4*sqrt(2))
#pragma unroll
  for (int i = 0; i < 16; ++i)
#pragma unroll
    for (int j = 0; j < 16; ++j) dst[17 + i * 16 + j] = (__bf16)(v[i] * v[j] * s);
#pragma unroll
  for (int pp = 273; pp < FP_; ++pp) dst[pp] = (__bf16)0.0f;
}

// ---------------------------------------------------------------------------
// v reshape+transpose: (B*Ntok, H*64) f32 -> (B,H,64,Ntok) bf16.
__global__ void vtrans_kernel(const float* __restrict__ g, __bf16* __restrict__ vt,
                              int Ntok) {
  size_t idx = (size_t)blockIdx.x * blockDim.x + threadIdx.x;
  size_t total = (size_t)B_ * Ntok * H_ * HD_;
  if (idx >= total) return;
  int c = (int)(idx & (size_t)(H_ * HD_ - 1));
  size_t m = idx >> 10;
  int h = c >> 6, e = c & 63;
  int b = (int)(m / Ntok), n = (int)(m % Ntok);
  vt[((size_t)(b * H_ + h) * HD_ + e) * Ntok + n] = (__bf16)g[idx];
}

// ---------------------------------------------------------------------------
// Attention. Block = 8 waves x 16 query rows = 128 queries. All waves share
// key tiles (32 keys x 288 feats + 32x64 v) staged into double-buffered LDS
// with async global->LDS DMA.  Per 16-key subtile: 9 WMMA for S = q_f k_f^T,
// causal mask, LDS bounce to A-layout, 4 WMMA for y += S @ V.  Row sums of
// masked S give the normalizer z = 1/(q_f . k_state + eps).
__global__ __launch_bounds__(256) void attn_kernel(
    const __bf16* __restrict__ qf, const __bf16* __restrict__ kf,
    const __bf16* __restrict__ kef, const __bf16* __restrict__ vt,
    const __bf16* __restrict__ vet, __bf16* __restrict__ attn_out) {
  __shared__ __align__(16) __bf16 sK[2][32 * FP_];  // 2 x 18 KB key-feature tile
  __shared__ __align__(16) __bf16 sV[2][64 * 32];   // 2 x 4 KB v tile (HD x keys)
  __shared__ __align__(16) __bf16 sS[8][16 * 32];   // per-wave S relayout scratch

  int tid = (int)threadIdx.x;
  int bh = blockIdx.x >> 4;    // b*H + h
  int qblk = blockIdx.x & 15;  // 16 query blocks of 128
  int wave = tid >> 5;
  int lane = tid & 31;
  int half = lane >> 4, col = lane & 15;
  int qb = qblk * 128 + wave * 16;  // this wave's query tile base

  const __bf16* qbase = qf + ((size_t)bh * N_ + qb) * FP_;
  const __bf16* kbase = kf + (size_t)bh * N_ * FP_;
  const __bf16* kebase = kef + (size_t)bh * ENC_ * FP_;
  const __bf16* vbase = vt + (size_t)bh * HD_ * N_;
  const __bf16* vebase = vet + (size_t)bh * HD_ * ENC_;

  // Stage one 32-key tile (k-features are 32*288 contiguous; v gathered rows).
  auto stage = [&](int buf, const __bf16* ksrc, const __bf16* vsrcBase, int kb,
                   int vld) {
    for (int c = tid; c < (32 * FP_) / 8; c += 256)
      cp16(&sK[buf][c * 8], ksrc + (size_t)c * 8);
    int row = tid >> 2, ch = (tid & 3) * 8;
    cp16(&sV[buf][row * 32 + ch], vsrcBase + (size_t)row * vld + kb + ch);
  };

  // q_f fragments for this wave's 16 rows, K = 288 (9 x 32), held in regs.
  v16bf qa[9];
#pragma unroll
  for (int t = 0; t < 9; ++t) qa[t] = load_a_rm(qbase + t * 32, FP_);

  v8f yacc[4];
#pragma unroll
  for (int j = 0; j < 4; ++j) yacc[j] = zero8();
  v8f rs = zero8();

  auto tile_compute = [&](const __bf16* kT, const __bf16* vT, int kbGlob,
                          bool maskOn) {
#pragma unroll
    for (int s = 0; s < 2; ++s) {
      int ksub = kbGlob + s * 16;
      if (maskOn && ksub > qb + 15) {  // whole subtile beyond diagonal
#pragma unroll
        for (int r = 0; r < 8; ++r)
          sS[wave][(8 * half + r) * 32 + s * 16 + col] = (__bf16)0.0f;
        continue;
      }
      v8f c = zero8();
      const __bf16* kp = kT + s * 16 * FP_;
#pragma unroll
      for (int t = 0; t < 9; ++t)
        c = wmma_bf16(qa[t], load_b_tr(kp + t * 32, FP_), c);
      if (maskOn) {
        int kg = ksub + col;
#pragma unroll
        for (int r = 0; r < 8; ++r)
          if (kg > qb + 8 * half + r) c[r] = 0.f;
      }
      rs += c;
#pragma unroll
      for (int r = 0; r < 8; ++r)
        sS[wave][(8 * half + r) * 32 + s * 16 + col] = (__bf16)c[r];
    }
    v16bf sa = load_a_rm(&sS[wave][0], 32);
#pragma unroll
    for (int j = 0; j < 4; ++j)
      yacc[j] = wmma_bf16(sa, load_b_tr(vT + j * 16 * 32, 32), yacc[j]);
  };

  // ---- causal self part: block-uniform tile loop, wave-local extent ----
  int nT = qblk * 4 + 4;   // tiles needed by the last wave of this block
  int nW = qb / 32 + 1;    // tiles this wave actually consumes
  stage(0, kbase, vbase, 0, N_);
  for (int kt = 0; kt < nT; ++kt) {
    int cur = kt & 1;
    if (kt + 1 < nT) {
      stage(cur ^ 1, kbase + (size_t)(kt + 1) * 32 * FP_, vbase, (kt + 1) * 32, N_);
      wait_async<6>();  // drain tile kt (in-order), keep tile kt+1 in flight
    } else {
      wait_async<0>();
    }
    __syncthreads();
    if (kt < nW) {
      int kb = kt * 32;
      tile_compute(&sK[cur][0], &sV[cur][0], kb, (kb + 31) > qb);
    }
    __syncthreads();
  }

  // ---- encoder prefix part (unmasked, 16 tiles) ----
  stage(0, kebase, vebase, 0, ENC_);
  for (int kt = 0; kt < ENC_ / 32; ++kt) {
    int cur = kt & 1;
    if (kt + 1 < ENC_ / 32) {
      stage(cur ^ 1, kebase + (size_t)(kt + 1) * 32 * FP_, vebase, (kt + 1) * 32,
            ENC_);
      wait_async<6>();
    } else {
      wait_async<0>();
    }
    __syncthreads();
    tile_compute(&sK[cur][0], &sV[cur][0], kt * 32, false);
    __syncthreads();
  }

  // ---- reduce row sums across the 16 lanes of each half-wave ----
#pragma unroll
  for (int mlt = 1; mlt < 16; mlt <<= 1) {
#pragma unroll
    for (int r = 0; r < 8; ++r) rs[r] += __shfl_xor(rs[r], mlt, 32);
  }

  // ---- scale by z and store bf16 in (B*N, H*HD) layout for the final GEMM ----
  int b = bh >> 4, h = bh & 15;
#pragma unroll
  for (int j = 0; j < 4; ++j) {
#pragma unroll
    for (int r = 0; r < 8; ++r) {
      float z = 1.0f / (rs[r] + 1e-12f);
      size_t row = (size_t)b * N_ + qb + 8 * half + r;
      attn_out[row * (H_ * HD_) + h * HD_ + j * 16 + col] =
          (__bf16)(yacc[j][r] * z);
    }
  }
}

// ---------------------------------------------------------------------------
extern "C" void kernel_launch(void* const* d_in, const int* in_sizes, int n_in,
                              void* d_out, int out_size, void* d_ws, size_t ws_size,
                              hipStream_t stream) {
  (void)in_sizes; (void)n_in; (void)out_size; (void)ws_size;
  const float* x    = (const float*)d_in[0];
  const float* Wq   = (const float*)d_in[1];
  const float* Wk   = (const float*)d_in[2];
  const float* Wv   = (const float*)d_in[3];
  const float* Wke  = (const float*)d_in[4];
  const float* Wve  = (const float*)d_in[5];
  const float* Wout = (const float*)d_in[6];
  float* out = (float*)d_out;

  char* p = (char*)d_ws;
  auto take = [&](size_t bytes) -> void* {
    void* r = (void*)p;
    p += (bytes + 255) & ~(size_t)255;
    return r;
  };

  const size_t BN = (size_t)B_ * N_;    // 4096 tokens
  const size_t BE = (size_t)B_ * ENC_;  // 1024 encoder tokens

  __bf16* x_bf   = (__bf16*)take(BN * D_ * 2);
  __bf16* Wq_bf  = (__bf16*)take((size_t)H_ * FD_ * D_ * 2);
  __bf16* Wk_bf  = (__bf16*)take((size_t)H_ * FD_ * D_ * 2);
  __bf16* Wv_bf  = (__bf16*)take((size_t)H_ * HD_ * D_ * 2);
  __bf16* Wke_bf = (__bf16*)take((size_t)H_ * FD_ * D_ * 2);
  __bf16* Wve_bf = (__bf16*)take((size_t)H_ * HD_ * D_ * 2);
  __bf16* Wo_bf  = (__bf16*)take((size_t)D_ * H_ * HD_ * 2);
  float*  qg     = (float*)take(BN * H_ * FD_ * 4);
  float*  kg     = (float*)take(BN * H_ * FD_ * 4);
  float*  vg     = (float*)take(BN * H_ * HD_ * 4);
  float*  keg    = (float*)take(BE * H_ * FD_ * 4);
  float*  veg    = (float*)take(BE * H_ * HD_ * 4);
  __bf16* qf     = (__bf16*)take((size_t)B_ * H_ * N_ * FP_ * 2);
  __bf16* kfb    = (__bf16*)take((size_t)B_ * H_ * N_ * FP_ * 2);
  __bf16* kef    = (__bf16*)take((size_t)B_ * H_ * ENC_ * FP_ * 2);
  __bf16* vt     = (__bf16*)take((size_t)B_ * H_ * HD_ * N_ * 2);
  __bf16* vet    = (__bf16*)take((size_t)B_ * H_ * HD_ * ENC_ * 2);
  __bf16* attn   = (__bf16*)take(BN * H_ * HD_ * 2);

  auto cvt = [&](const float* src, __bf16* dst, size_t n) {
    f2bf_kernel<<<(unsigned)((n + 255) / 256), 256, 0, stream>>>(src, dst, (int)n);
  };
  cvt(x, x_bf, BN * D_);
  cvt(Wq, Wq_bf, (size_t)H_ * FD_ * D_);
  cvt(Wk, Wk_bf, (size_t)H_ * FD_ * D_);
  cvt(Wv, Wv_bf, (size_t)H_ * HD_ * D_);
  cvt(Wke, Wke_bf, (size_t)H_ * FD_ * D_);
  cvt(Wve, Wve_bf, (size_t)H_ * HD_ * D_);
  cvt(Wout, Wo_bf, (size_t)D_ * H_ * HD_);

  const int BIGSEG = 1 << 28;  // identity row remap
  gemm_xwt<<<dim3(1, (unsigned)(BN / 64)), 256, 0, stream>>>(
      x_bf, Wq_bf, qg, (int)BN, H_ * FD_, D_, BIGSEG, 0);
  gemm_xwt<<<dim3(1, (unsigned)(BN / 64)), 256, 0, stream>>>(
      x_bf, Wk_bf, kg, (int)BN, H_ * FD_, D_, BIGSEG, 0);
  gemm_xwt<<<dim3(4, (unsigned)(BN / 64)), 256, 0, stream>>>(
      x_bf, Wv_bf, vg, (int)BN, H_ * HD_, D_, BIGSEG, 0);
  gemm_xwt<<<dim3(1, (unsigned)(BE / 64)), 256, 0, stream>>>(
      x_bf, Wke_bf, keg, (int)BE, H_ * FD_, D_, ENC_, N_);
  gemm_xwt<<<dim3(4, (unsigned)(BE / 64)), 256, 0, stream>>>(
      x_bf, Wve_bf, veg, (int)BE, H_ * HD_, D_, ENC_, N_);

  {
    int tq = (int)(BN * H_);
    feature_kernel<<<(tq + 255) / 256, 256, 0, stream>>>(qg, qf, N_);
    feature_kernel<<<(tq + 255) / 256, 256, 0, stream>>>(kg, kfb, N_);
    int te = (int)(BE * H_);
    feature_kernel<<<(te + 255) / 256, 256, 0, stream>>>(keg, kef, ENC_);
  }
  vtrans_kernel<<<(unsigned)((BN * H_ * HD_ + 255) / 256), 256, 0, stream>>>(vg, vt, N_);
  vtrans_kernel<<<(unsigned)((BE * H_ * HD_ + 255) / 256), 256, 0, stream>>>(veg, vet, ENC_);

  attn_kernel<<<B_ * H_ * (N_ / 128), 256, 0, stream>>>(qf, kfb, kef, vt, vet, attn);

  gemm_xwt<<<dim3(4, (unsigned)(BN / 64)), 256, 0, stream>>>(
      attn, Wo_bf, out, (int)BN, H_ * HD_, D_, BIGSEG, 0);
}
